// CausalAttention_34067680592114
// MI455X (gfx1250) — matrix operand order repaired and checked by reading, not compiled, not run on hardware
//
#include <hip/hip_runtime.h>
#include <hip/hip_bf16.h>

// ---------------------------------------------------------------------------
// CDNA5 (gfx1250) causal attention: bf16 WMMA GEMMs + flash-attention core
// with async global->LDS double buffering (ASYNCcnt) where available.
// Wave32 everywhere. Matrix ops via v_wmma_f32_16x16x32_bf16.
// ---------------------------------------------------------------------------

typedef __attribute__((ext_vector_type(16))) __bf16 v16bf;
typedef __attribute__((ext_vector_type(8)))  __bf16 v8bf;
typedef __attribute__((ext_vector_type(8)))  float  v8f;

// Problem constants (match reference)
constexpr int Bc   = 2;
constexpr int Sc   = 2048;
constexpr int HIDc = 2048;
constexpr int Hc   = 16;
constexpr int Dc   = 128;           // head dim
constexpr int Mc   = Bc * Sc;       // 4096 tokens
constexpr int Kc   = HIDc;          // GEMM reduction dim
constexpr int Nc   = HIDc;          // GEMM output dim

static __device__ inline v16bf concat8(v8bf lo, v8bf hi) {
  return __builtin_shufflevector(lo, hi, 0,1,2,3,4,5,6,7,8,9,10,11,12,13,14,15);
}

// A-fragment (16x32 bf16, MxK): per ISA 7.12.2, lane&15 = row M; lanes 0-15
// hold K = {0..7, 16..23}, lanes 16-31 hold K = {8..15, 24..31}.
// p = &A[row*lda + kbase + (lane<16 ? 0 : 8)]
static __device__ inline v16bf load_a_frag(const __bf16* __restrict__ p) {
  v8bf lo = *reinterpret_cast<const v8bf*>(p);
  v8bf hi = *reinterpret_cast<const v8bf*>(p + 16);
  return concat8(lo, hi);
}

// B-fragment (32x16 bf16, KxN): lane&15 = column N; lanes 0-15 hold
// K = {0..7, 16..23}, lanes 16-31 hold K = {8..15, 24..31}.
static __device__ inline v16bf load_b_frag(const __bf16* __restrict__ p) {
  v8bf lo = *reinterpret_cast<const v8bf*>(p);
  v8bf hi = *reinterpret_cast<const v8bf*>(p + 16);
  return concat8(lo, hi);
}

static __device__ inline v8f wmma_bf16(v16bf a, v16bf b, v8f c) {
  // (neg_a, A, neg_b, B, c_mod, C, reuse_a, reuse_b)
  return __builtin_amdgcn_wmma_f32_16x16x32_bf16(false, a, false, b, (short)0, c,
                                                 false, false);
}

// ---- async global->LDS copy (16B per lane), with synchronous fallback ------
#if defined(__AMDGCN__) && __has_builtin(__builtin_amdgcn_global_load_async_to_lds_b128)
#define FA_ASYNC 1
typedef __attribute__((__vector_size__(16))) int v4i_t;
typedef __attribute__((address_space(1))) v4i_t as1_v4i;
typedef __attribute__((address_space(3))) v4i_t as3_v4i;
#define COPY16(gsrc, ldst)                                                     \
  __builtin_amdgcn_global_load_async_to_lds_b128(                              \
      (as1_v4i*)(gsrc), (as3_v4i*)(ldst), 0, 0)
#define STAGE_WAIT_ALL()  asm volatile("s_wait_asynccnt 0x0" ::: "memory")
#define STAGE_WAIT_HALF() asm volatile("s_wait_asynccnt 0x20" ::: "memory")
#else
#define FA_ASYNC 0
#define COPY16(gsrc, ldst) (*(v8bf*)(ldst) = *(const v8bf*)(gsrc))
#define STAGE_WAIT_ALL()  asm volatile("s_wait_dscnt 0x0" ::: "memory")
#define STAGE_WAIT_HALF() asm volatile("s_wait_dscnt 0x0" ::: "memory")
#endif

// ---------------------------------------------------------------------------
// Stage 1: fp32 -> bf16 convert
// ---------------------------------------------------------------------------
__global__ void f32_to_bf16_kernel(const float* __restrict__ in,
                                   __bf16* __restrict__ out, int n) {
  int i = blockIdx.x * blockDim.x + threadIdx.x;
  if (i < n) out[i] = (__bf16)in[i];
}

// ---------------------------------------------------------------------------
// Stage 2/5: bf16 GEMM  y[m,n] = sum_k A[m,k] * W[n,k]   (i.e. x @ W^T)
// Each wave computes a 32x32 tile; block = 4 waves = 64x64 tile.
// MODE 0: store bf16 row-major [M,N]
// MODE 1: store bf16 per-head transposed: Vt[((b*H+h)*D+d)*S + s]
// MODE 2: store fp32 row-major [M,N]
// ---------------------------------------------------------------------------
template <int MODE>
__global__ __launch_bounds__(128)
void gemm_bf16_kernel(const __bf16* __restrict__ A,
                      const __bf16* __restrict__ W,
                      void* __restrict__ out) {
  const int lane = threadIdx.x & 31;
  const int wave = threadIdx.x >> 5;
  const int wx = wave & 1, wy = wave >> 1;
  const int l15 = lane & 15;
  const int hl  = lane >> 4;            // 0: lanes 0-15, 1: lanes 16-31

  const int n0 = blockIdx.x * 64 + wx * 32;
  const int m0 = blockIdx.y * 64 + wy * 32;

  const __bf16* arow0 = A + (size_t)(m0 + l15) * Kc + hl * 8;
  const __bf16* arow1 = arow0 + (size_t)16 * Kc;
  const __bf16* brow0 = W + (size_t)(n0 + l15) * Kc + hl * 8;
  const __bf16* brow1 = brow0 + (size_t)16 * Kc;

  v8f acc00 = {}, acc01 = {}, acc10 = {}, acc11 = {};

  for (int k = 0; k < Kc; k += 32) {
    // stream-prefetch the B operand a few tiles ahead (global_prefetch_b8)
    if (k + 256 < Kc) __builtin_prefetch(brow0 + k + 256, 0, 0);

    v16bf a0 = load_a_frag(arow0 + k);
    v16bf a1 = load_a_frag(arow1 + k);
    v16bf b0 = load_b_frag(brow0 + k);
    v16bf b1 = load_b_frag(brow1 + k);

    acc00 = wmma_bf16(a0, b0, acc00);
    acc01 = wmma_bf16(a0, b1, acc01);
    acc10 = wmma_bf16(a1, b0, acc10);
    acc11 = wmma_bf16(a1, b1, acc11);
  }

  // C/D layout: VGPR i -> row (i + hl*8) of the 16x16 tile, col = l15.
#pragma unroll
  for (int mi = 0; mi < 2; ++mi) {
#pragma unroll
    for (int ni = 0; ni < 2; ++ni) {
      const v8f& acc = (mi == 0) ? (ni == 0 ? acc00 : acc01)
                                 : (ni == 0 ? acc10 : acc11);
#pragma unroll
      for (int i = 0; i < 8; ++i) {
        int m = m0 + mi * 16 + i + hl * 8;
        int n = n0 + ni * 16 + l15;
        float v = acc[i];
        if (MODE == 0) {
          ((__bf16*)out)[(size_t)m * Nc + n] = (__bf16)v;
        } else if (MODE == 1) {
          int b = m / Sc, s = m % Sc;
          int h = n / Dc, d = n % Dc;
          ((__bf16*)out)[(((size_t)(b * Hc + h) * Dc + d) * Sc) + s] = (__bf16)v;
        } else {
          ((float*)out)[(size_t)m * Nc + n] = v;
        }
      }
    }
  }
}

// ---------------------------------------------------------------------------
// Stage 3: RoPE on Q and K (in place, bf16 storage, fp32 math)
// freqs_cis: [S, D/2, 2] fp32 (cos, sin)
// ---------------------------------------------------------------------------
__global__ void rope_kernel(__bf16* __restrict__ Q, __bf16* __restrict__ Kb,
                            const float* __restrict__ fc) {
  int idx = blockIdx.x * blockDim.x + threadIdx.x;
  constexpr int total = Bc * Sc * Hc * (Dc / 2);
  if (idx >= total) return;
  int j = idx & (Dc / 2 - 1);
  int t = idx >> 6;                 // / (D/2)
  int h = t & (Hc - 1);
  t >>= 4;                          // / H
  int s = t & (Sc - 1);
  int b = t >> 11;                  // / S
  size_t base = ((size_t)(b * Sc + s)) * HIDc + h * Dc + 2 * j;
  float c  = fc[((size_t)s * (Dc / 2) + j) * 2 + 0];
  float si = fc[((size_t)s * (Dc / 2) + j) * 2 + 1];

  float qe = (float)Q[base], qo = (float)Q[base + 1];
  Q[base]     = (__bf16)(qe * c - qo * si);
  Q[base + 1] = (__bf16)(qe * si + qo * c);

  float ke = (float)Kb[base], ko = (float)Kb[base + 1];
  Kb[base]     = (__bf16)(ke * c - ko * si);
  Kb[base + 1] = (__bf16)(ke * si + ko * c);
}

// ---------------------------------------------------------------------------
// Stage 4: flash attention. One wave (32 threads) per 32-query tile per (b,h).
// K/V chunks (32 keys) staged in LDS with async double buffering; QK^T and
// P*V via v_wmma_f32_16x16x32_bf16 (32 WMMAs per key chunk). fp32 softmax.
// ---------------------------------------------------------------------------
__global__ __launch_bounds__(32)
void flash_attn_kernel(const __bf16* __restrict__ Q,
                       const __bf16* __restrict__ Kb,
                       const __bf16* __restrict__ Vt,   // [B,H,D,S]
                       __bf16* __restrict__ O) {
  __shared__ __bf16 kbuf[2][32 * 128];   // 32 keys x 128 headdim (row = key)
  __shared__ __bf16 vbuf[2][128 * 32];   // 128 headdim x 32 keys (row = d)
  __shared__ __bf16 pbuf[32 * 32];       // P tile: 32 q rows x 32 key cols

  const int lane = threadIdx.x & 31;
  const int l15  = lane & 15;
  const int hl   = lane >> 4;

  const int bh = blockIdx.x;
  const int b  = bh / Hc, h = bh % Hc;
  const int q0 = blockIdx.y * 32;
  const int d0 = h * Dc;

  constexpr float sm_scale = 0.08838834764831845f;  // 1/sqrt(128)

  // Preload Q fragments: 2 query sub-tiles x 4 head-dim chunks of 32
  v16bf qf[2][4];
#pragma unroll
  for (int mt = 0; mt < 2; ++mt) {
    const __bf16* qrow =
        Q + ((size_t)(b * Sc) + q0 + mt * 16 + l15) * HIDc + d0 + hl * 8;
#pragma unroll
    for (int dc = 0; dc < 4; ++dc) qf[mt][dc] = load_a_frag(qrow + dc * 32);
  }

  v8f o[2][8];
  float m_r[2][8], l_r[2][8];
#pragma unroll
  for (int mt = 0; mt < 2; ++mt)
#pragma unroll
    for (int i = 0; i < 8; ++i) {
      o[mt][i] = (v8f){0.f,0.f,0.f,0.f,0.f,0.f,0.f,0.f};
      m_r[mt][i] = -1e30f;
      l_r[mt][i] = 0.f;
    }

  // stage one 32-key chunk of K and V into LDS slot (16B per lane per issue)
  auto stage = [&](int kt, int slot) {
    const size_t krowg = (size_t)(b * Sc + kt * 32);
#pragma unroll
    for (int j = 0; j < 16; ++j) {
      const __bf16* g = Kb + (krowg + 2 * j + hl) * HIDc + d0 + l15 * 8;
      __bf16* l = &kbuf[slot][(2 * j + hl) * 128 + l15 * 8];
      COPY16(g, l);
    }
    const size_t vbase = (size_t)(b * Hc + h) * Dc;
#pragma unroll
    for (int j = 0; j < 16; ++j) {
      int row = 8 * j + (lane >> 2);
      const __bf16* g = Vt + (vbase + row) * Sc + kt * 32 + (lane & 3) * 8;
      __bf16* l = &vbuf[slot][row * 32 + (lane & 3) * 8];
      COPY16(g, l);
    }
  };

  const int nkt = q0 / 32 + 1;    // 32-key chunks covering the causal span

#if FA_ASYNC
  stage(0, 0);
#endif

  for (int kt = 0; kt < nkt; ++kt) {
    int slot;
#if FA_ASYNC
    slot = kt & 1;
    if (kt + 1 < nkt) {
      stage(kt + 1, slot ^ 1);   // stream next chunk behind the compute
      STAGE_WAIT_HALF();         // ASYNCcnt <= 32 -> chunk kt resident
    } else {
      STAGE_WAIT_ALL();
    }
#else
    slot = 0;
    stage(kt, 0);
    STAGE_WAIT_ALL();
#endif
    const __bf16* kb = kbuf[slot];
    const __bf16* vb = vbuf[slot];

    // ---- scores: 2 q-subtiles x 2 key-subtiles, 4 WMMAs each over D=128 ----
    v8f sc[2][2];
#pragma unroll
    for (int mt = 0; mt < 2; ++mt)
#pragma unroll
      for (int sub = 0; sub < 2; ++sub)
        sc[mt][sub] = (v8f){0.f,0.f,0.f,0.f,0.f,0.f,0.f,0.f};

#pragma unroll
    for (int sub = 0; sub < 2; ++sub) {
#pragma unroll
      for (int dc = 0; dc < 4; ++dc) {
        v16bf kf = load_b_frag(&kb[(sub * 16 + l15) * 128 + dc * 32 + hl * 8]);
        sc[0][sub] = wmma_bf16(qf[0][dc], kf, sc[0][sub]);
        sc[1][sub] = wmma_bf16(qf[1][dc], kf, sc[1][sub]);
      }
    }

    // ---- scale + causal mask + online softmax per q-subtile ----
#pragma unroll
    for (int mt = 0; mt < 2; ++mt) {
#pragma unroll
      for (int sub = 0; sub < 2; ++sub) {
#pragma unroll
        for (int i = 0; i < 8; ++i) {
          int row = q0 + mt * 16 + i + hl * 8;
          int col = kt * 32 + sub * 16 + l15;
          float v = sc[mt][sub][i] * sm_scale;
          sc[mt][sub][i] = (col > row) ? -1e30f : v;
        }
      }
#pragma unroll
      for (int i = 0; i < 8; ++i) {
        float mx = fmaxf(sc[mt][0][i], sc[mt][1][i]);
#pragma unroll
        for (int off = 1; off < 16; off <<= 1)
          mx = fmaxf(mx, __shfl_xor(mx, off, 32));
        float mnew  = fmaxf(m_r[mt][i], mx);
        float alpha = __expf(m_r[mt][i] - mnew);
        float p0 = __expf(sc[mt][0][i] - mnew);
        float p1 = __expf(sc[mt][1][i] - mnew);
        float rs = p0 + p1;
#pragma unroll
        for (int off = 1; off < 16; off <<= 1)
          rs += __shfl_xor(rs, off, 32);
        l_r[mt][i] = l_r[mt][i] * alpha + rs;
        m_r[mt][i] = mnew;
#pragma unroll
        for (int nc = 0; nc < 8; ++nc) o[mt][nc][i] *= alpha;
        // stage P (C-layout) to LDS for transpose into A-layout
        pbuf[(mt * 16 + i + hl * 8) * 32 + l15]      = (__bf16)p0;
        pbuf[(mt * 16 + i + hl * 8) * 32 + 16 + l15] = (__bf16)p1;
      }
    }

    // single-wave block: DS ops are in-order; explicit wait for store->load
    asm volatile("s_wait_dscnt 0x0" ::: "memory");

    v16bf pf0 = load_a_frag(&pbuf[(l15) * 32 + hl * 8]);
    v16bf pf1 = load_a_frag(&pbuf[(16 + l15) * 32 + hl * 8]);

    // ---- O += P @ V : 16 WMMAs across D=128 ----
#pragma unroll
    for (int nc = 0; nc < 8; ++nc) {
      v16bf vf = load_b_frag(&vb[(nc * 16 + l15) * 32 + hl * 8]);
      o[0][nc] = wmma_bf16(pf0, vf, o[0][nc]);
      o[1][nc] = wmma_bf16(pf1, vf, o[1][nc]);
    }
    asm volatile("s_wait_dscnt 0x0" ::: "memory");
  }

  // ---- normalize and store per-head output (row-major [B,S,HID] bf16) ----
#pragma unroll
  for (int mt = 0; mt < 2; ++mt) {
#pragma unroll
    for (int i = 0; i < 8; ++i) {
      float inv = 1.0f / l_r[mt][i];
      size_t rbase =
          ((size_t)(b * Sc) + q0 + mt * 16 + i + hl * 8) * HIDc + d0;
#pragma unroll
      for (int nc = 0; nc < 8; ++nc) {
        O[rbase + nc * 16 + l15] = (__bf16)(o[mt][nc][i] * inv);
      }
    }
  }
}

// ---------------------------------------------------------------------------
// Host-side orchestration
// ---------------------------------------------------------------------------
extern "C" void kernel_launch(void* const* d_in, const int* in_sizes, int n_in,
                              void* d_out, int out_size, void* d_ws, size_t ws_size,
                              hipStream_t stream) {
  const float* x  = (const float*)d_in[0];
  const float* fc = (const float*)d_in[1];
  const float* wq = (const float*)d_in[2];
  const float* wk = (const float*)d_in[3];
  const float* wv = (const float*)d_in[4];
  const float* wo = (const float*)d_in[5];

  char* ws = (char*)d_ws;
  size_t off = 0;
  auto carve = [&](size_t bytes) {
    void* p = ws + off;
    off += (bytes + 255) & ~(size_t)255;
    return p;
  };

  __bf16* Xbf = (__bf16*)carve((size_t)Mc * Kc * 2);
  __bf16* Wq  = (__bf16*)carve((size_t)Nc * Kc * 2);
  __bf16* Wk  = (__bf16*)carve((size_t)Nc * Kc * 2);
  __bf16* Wv  = (__bf16*)carve((size_t)Nc * Kc * 2);
  __bf16* Wo  = (__bf16*)carve((size_t)Nc * Kc * 2);
  __bf16* Qbf = (__bf16*)carve((size_t)Mc * HIDc * 2);
  __bf16* Kbf = (__bf16*)carve((size_t)Mc * HIDc * 2);
  __bf16* Vt  = (__bf16*)carve((size_t)Bc * Hc * Dc * Sc * 2);
  __bf16* Obf = (__bf16*)carve((size_t)Mc * HIDc * 2);

  // Stage 1: converts
  {
    int n = Mc * Kc;
    f32_to_bf16_kernel<<<(n + 255) / 256, 256, 0, stream>>>(x, Xbf, n);
    n = Nc * Kc;
    f32_to_bf16_kernel<<<(n + 255) / 256, 256, 0, stream>>>(wq, Wq, n);
    f32_to_bf16_kernel<<<(n + 255) / 256, 256, 0, stream>>>(wk, Wk, n);
    f32_to_bf16_kernel<<<(n + 255) / 256, 256, 0, stream>>>(wv, Wv, n);
    f32_to_bf16_kernel<<<(n + 255) / 256, 256, 0, stream>>>(wo, Wo, n);
  }

  dim3 ggrid(Nc / 64, Mc / 64);
  // Stage 2: QKV projections
  gemm_bf16_kernel<0><<<ggrid, 128, 0, stream>>>(Xbf, Wq, Qbf);
  gemm_bf16_kernel<0><<<ggrid, 128, 0, stream>>>(Xbf, Wk, Kbf);
  gemm_bf16_kernel<1><<<ggrid, 128, 0, stream>>>(Xbf, Wv, Vt);

  // Stage 3: RoPE on Q and K
  {
    int n = Bc * Sc * Hc * (Dc / 2);
    rope_kernel<<<(n + 255) / 256, 256, 0, stream>>>(Qbf, Kbf, fc);
  }

  // Stage 4: flash attention (32-query tile per wave)
  flash_attn_kernel<<<dim3(Bc * Hc, Sc / 32), 32, 0, stream>>>(Qbf, Kbf, Vt, Obf);

  // Stage 5: output projection -> fp32 d_out
  gemm_bf16_kernel<2><<<ggrid, 128, 0, stream>>>(Obf, Wo, (float*)d_out);
}
